// Net_26783416057886
// MI455X (gfx1250) — matrix operand order repaired and checked by reading, not compiled, not run on hardware
//
#include <hip/hip_runtime.h>
#include <hip/hip_bf16.h>

typedef __attribute__((ext_vector_type(16))) _Float16 v16h;
typedef __attribute__((ext_vector_type(8)))  _Float16 v8h;
typedef __attribute__((ext_vector_type(8)))  float    v8f;

#define DEV __device__ __forceinline__

DEV v16h hzero16() {
  v16h r;
#pragma unroll
  for (int j = 0; j < 16; ++j) r[j] = (_Float16)0.f;
  return r;
}
DEV v8f fzero8() {
  v8f r;
#pragma unroll
  for (int j = 0; j < 8; ++j) r[j] = 0.f;
  return r;
}

DEV v8f wmma_f16(v16h a, v16h b, v8f c) {
  // D(f32 16x16) = A(f16 16x32) * B(f16 32x16) + C
  return __builtin_amdgcn_wmma_f32_16x16x32_f16(
      /*neg_a=*/false, a, /*neg_b=*/false, b,
      /*c_mod=*/(short)0, c, /*reuse_a=*/false, /*reuse_b=*/false);
}

// A-fragment (16x32 f16): lane m = L%16; halfs 0..15 -> K = 2*khi..2*khi+15,
// khi = 8*(L/16). `row` points at (weights row + kbase).
DEV v16h load_a_frag(const _Float16* row, int khi) {
  v8h lo = *(const v8h*)(row + 2 * khi);
  v8h hi = *(const v8h*)(row + 2 * khi + 8);
  v16h r;
#pragma unroll
  for (int j = 0; j < 8; ++j) { r[j] = lo[j]; r[j + 8] = hi[j]; }
  return r;
}

// ---------------------------------------------------------------------------
// Generic WMMA GEMM / conv kernel, fully specialized on shape.
// src: f16 NHWC [Bn,H,W,Cin]; wt: f16 [NPOS][Cout][Cin]; dst: f16 NHWC [*,Cout]
// NPOS==9 -> 3x3 pad-1 conv (pos = ky*3+kx); NPOS==1 -> 1x1 / token GEMM.
// bias f32[Cout] optional; mask f16[N] optional (after bias);
// resid f16[N,Cout] optional (after mask); act: 0 none, 1 relu, 2 gelu.
// block = 256 (8 waves). MT = Cout/16 m-tiles. For MT<8 the waves split into
// 8/MT independent 16-column groups (all waves busy); for MT>=8 each wave
// iterates m-tiles wid, wid+8, ... All loops compile-time -> WMMA at full EXEC.
// grid.x = N / (16 * NG).
// ---------------------------------------------------------------------------
template <int MT, int KC, int NPOS>
__global__ void __launch_bounds__(256) conv_gemm_wmma(
    const _Float16* __restrict__ src, const _Float16* __restrict__ wt,
    _Float16* __restrict__ dst, const float* __restrict__ bias,
    const _Float16* __restrict__ mask, const _Float16* __restrict__ resid,
    int Bn, int H, int W, int act)
{
  constexpr int Cout = MT * 16;
  constexpr int Cin  = KC * 32;
  constexpr int MTW  = (MT < 8) ? MT : 8;   // m-tiles handled per column group
  constexpr int NG   = 8 / MTW;             // column groups per workgroup
  constexpr int ITER = (MT + 7) / 8;        // m-tiles per wave

  const int tid = threadIdx.x;
  const int lane = tid & 31, wid = tid >> 5;
  const int nlo = lane & 15, khalf = lane >> 4;
  const int mslot = wid % MTW;
  const int grp = wid / MTW;
  const int hw = H * W;
  const int n = (blockIdx.x * NG + grp) * 16 + nlo;
  int b = 0, y = 0, x = 0;
  if (NPOS == 9) { b = n / hw; int r = n - b * hw; y = r / W; x = r - y * W; }

  v8f acc[ITER];
#pragma unroll
  for (int i = 0; i < ITER; ++i) acc[i] = fzero8();

#pragma unroll
  for (int pos = 0; pos < NPOS; ++pos) {
    long rowbase;
    bool valid = true;
    if (NPOS == 9) {
      int dy = pos / 3 - 1, dx = pos - (pos / 3) * 3 - 1;
      int yy = y + dy, xx = x + dx;
      valid = (yy >= 0) && (yy < H) && (xx >= 0) && (xx < W);
      rowbase = ((long)(b * H + yy) * W + xx) * Cin;
      // prefetch the next shifted row while this one is consumed
      if (pos + 1 < NPOS) {
        int dy2 = (pos + 1) / 3 - 1, dx2 = (pos + 1) - ((pos + 1) / 3) * 3 - 1;
        int y2 = y + dy2, x2 = x + dx2;
        if ((y2 >= 0) && (y2 < H) && (x2 >= 0) && (x2 < W))
          __builtin_prefetch(src + ((long)(b * H + y2) * W + x2) * Cin, 0, 0);
      }
    } else {
      rowbase = (long)n * Cin;
    }
    const _Float16* wp = wt + (long)pos * Cout * Cin;
#pragma unroll
    for (int kc = 0; kc < KC; ++kc) {
      v16h bf;
      if (valid) bf = *(const v16h*)(src + rowbase + kc * 32 + khalf * 16);
      else       bf = hzero16();
#pragma unroll
      for (int i = 0; i < ITER; ++i) {
        const int mt = mslot + i * 8;
        const _Float16* row = wp + (long)(mt * 16 + nlo) * Cin + kc * 32;
        v16h af = load_a_frag(row, khalf * 8);
        acc[i] = wmma_f16(af, bf, acc[i]);
      }
    }
  }

  __syncthreads();  // allow safe in-place (dst==src) 1x1 GEMMs within the WG

  float mval = 1.f;
  if (mask) mval = (float)mask[n];
#pragma unroll
  for (int i = 0; i < ITER; ++i) {
    const int mt = mslot + i * 8;
    int mbase = mt * 16 + khalf * 8;
    v8h ov;
#pragma unroll
    for (int v = 0; v < 8; ++v) {
      int m = mbase + v;
      float val = acc[i][v];
      if (bias)  val += bias[m];
      if (mask)  val *= mval;
      if (resid) val += (float)resid[(long)n * Cout + m];
      if (act == 1)      val = fmaxf(val, 0.f);
      else if (act == 2) val = 0.5f * val * (1.f + erff(val * 0.70710678118f));
      ov[v] = (_Float16)val;
    }
    *(v8h*)(dst + (long)n * Cout + mbase) = ov;
  }
}

// ---------------------------------------------------------------------------
// Transposed conv 2x2 stride 2 (non-overlapping): y[b,o,2h+p,2w+q] = sum_c x*w[c,o,p,q]
// wt: f16 [4(pq)][Cout][Cin]; dst: f16 NHWC [Bn,2H,2W,Cout]; mask indexed at output.
// 4*MT tiles spread statically over 8 waves.
// ---------------------------------------------------------------------------
template <int MT, int KC>
__global__ void __launch_bounds__(256) deconv2x2_wmma(
    const _Float16* __restrict__ src, const _Float16* __restrict__ wt,
    _Float16* __restrict__ dst, const _Float16* __restrict__ mask,
    int Bn, int H, int W)
{
  constexpr int Cout = MT * 16;
  constexpr int Cin  = KC * 32;
  constexpr int TILES = 4 * MT;
  constexpr int ITER  = (TILES + 7) / 8;

  const int tid = threadIdx.x;
  const int lane = tid & 31, wid = tid >> 5;
  const int nlo = lane & 15, khalf = lane >> 4;
  const int hw = H * W;
  const int n = blockIdx.x * 16 + nlo;
  const int b = n / hw; int r = n - b * hw; const int y = r / W, x = r - y * W;
  const int H2 = 2 * H, W2 = 2 * W;

#pragma unroll
  for (int it = 0; it < ITER; ++it) {
    const int t = wid + it * 8;
    const int pos = t / MT, mt = t - pos * MT;
    v8f acc = fzero8();
    const _Float16* wp = wt + (long)pos * Cout * Cin;
#pragma unroll
    for (int kc = 0; kc < KC; ++kc) {
      v16h bf = *(const v16h*)(src + (long)n * Cin + kc * 32 + khalf * 16);
      v16h af = load_a_frag(wp + (long)(mt * 16 + nlo) * Cin + kc * 32, khalf * 8);
      acc = wmma_f16(af, bf, acc);
    }
    int p = pos >> 1, q = pos & 1;
    long on = (long)(b * H2 + 2 * y + p) * W2 + (2 * x + q);
    float mval = mask ? (float)mask[on] : 1.f;
    int mbase = mt * 16 + khalf * 8;
    v8h ov;
#pragma unroll
    for (int v = 0; v < 8; ++v) ov[v] = (_Float16)(acc[v] * mval);
    *(v8h*)(dst + on * Cout + mbase) = ov;
  }
}

// ---------------------------------------------------------------------------
// Attention: one workgroup per (batch, head). tokens=49 padded to 64, d=32, H=4.
// qkv: f16 [B*49, 384] (q|k|v each 128 = 4 heads * 32). o: f16 [B*49, 128].
// ---------------------------------------------------------------------------
__global__ void __launch_bounds__(256) attn_kernel(
    const _Float16* __restrict__ qkv, _Float16* __restrict__ o)
{
  __shared__ alignas(64) _Float16 sq[64 * 32];
  __shared__ alignas(64) _Float16 sk[64 * 32];
  __shared__ alignas(64) _Float16 sv[64 * 32];
  __shared__ alignas(64) float    ssc[64 * 64];
  __shared__ alignas(64) _Float16 spr[64 * 64];

  const int tid = threadIdx.x;
  const int b = blockIdx.x >> 2, h = blockIdx.x & 3;

  for (int idx = tid; idx < 64 * 32; idx += 256) {
    int t = idx >> 5, d = idx & 31;
    _Float16 qv = (_Float16)0.f, kv = (_Float16)0.f, vv = (_Float16)0.f;
    if (t < 49) {
      long base = ((long)(b * 49 + t)) * 384 + h * 32 + d;
      qv = qkv[base]; kv = qkv[base + 128]; vv = qkv[base + 256];
    }
    sq[idx] = qv; sk[idx] = kv; sv[idx] = vv;
  }
  __syncthreads();

  const int lane = tid & 31, wid = tid >> 5;
  const int nlo = lane & 15, khalf = lane >> 4;
  const float scale = 0.17677669529663687f;  // 1/sqrt(32)

  // scores = q @ k^T * scale : M=64, N=64, K=32 -> 16 wave tiles
#pragma unroll
  for (int rep = 0; rep < 2; ++rep) {
    int t2 = wid + rep * 8;
    int mt = t2 >> 2, nt = t2 & 3;
    v16h af = load_a_frag(&sq[(mt * 16 + nlo) * 32], khalf * 8);
    v16h bf = *(const v16h*)&sk[(nt * 16 + nlo) * 32 + khalf * 16];
    v8f acc = fzero8();
    acc = wmma_f16(af, bf, acc);
    int mbase = mt * 16 + khalf * 8;
#pragma unroll
    for (int v = 0; v < 8; ++v) ssc[(mbase + v) * 64 + nt * 16 + nlo] = acc[v] * scale;
  }
  __syncthreads();

  if (tid < 64) {
    float mx = -1e30f;
    for (int j = 0; j < 49; ++j) mx = fmaxf(mx, ssc[tid * 64 + j]);
    float s = 0.f;
    for (int j = 0; j < 49; ++j) s += __expf(ssc[tid * 64 + j] - mx);
    float inv = 1.f / s;
    for (int j = 0; j < 64; ++j)
      spr[tid * 64 + j] = (_Float16)((j < 49) ? __expf(ssc[tid * 64 + j] - mx) * inv : 0.f);
  }
  __syncthreads();

  // out = P @ V : M=64, N=32, K=64 -> 8 wave tiles (one per wave), 2 K-chunks
  {
    int mt = wid >> 1, nt = wid & 1;
    v8f acc = fzero8();
#pragma unroll
    for (int kc = 0; kc < 2; ++kc) {
      v16h af = load_a_frag(&spr[(mt * 16 + nlo) * 64 + kc * 32], khalf * 8);
      v16h bf;
#pragma unroll
      for (int jj = 0; jj < 16; ++jj)
        bf[jj] = sv[(kc * 32 + khalf * 16 + jj) * 32 + nt * 16 + nlo];
      acc = wmma_f16(af, bf, acc);
    }
    int mbase = mt * 16 + khalf * 8;
#pragma unroll
    for (int v = 0; v < 8; ++v) {
      int i = mbase + v;
      if (i < 49)
        o[((long)(b * 49 + i)) * 128 + h * 32 + nt * 16 + nlo] = (_Float16)acc[v];
    }
  }
}

// --------------------------- small support kernels -------------------------

__global__ void zero_f32_kernel(float* __restrict__ p, int n) {
  int i = blockIdx.x * 256 + threadIdx.x;
  if (i < n) p[i] = 0.f;
}

// front: mask m0, ident = w_ds*x*m, t_a = conv3x3(x; 1->32)*m  (K=9, VALU)
__global__ void __launch_bounds__(256) front_kernel(
    const float* __restrict__ x, const float* __restrict__ wds,
    const float* __restrict__ wc1, _Float16* __restrict__ m0,
    _Float16* __restrict__ tid0, _Float16* __restrict__ ta, long total)
{
  for (long idx = (long)blockIdx.x * 256 + threadIdx.x; idx < total;
       idx += (long)gridDim.x * 256) {
    int r = (int)(idx % 784); long b = idx / 784;
    int y = r / 28, xx = r - y * 28;
    float xv = x[idx];
    float m = (fabsf(xv) > 0.f) ? 1.f : 0.f;
    m0[idx] = (_Float16)m;
    float acc[32];
#pragma unroll
    for (int c = 0; c < 32; ++c) acc[c] = 0.f;
    for (int ky = 0; ky < 3; ++ky) {
      int yy = y + ky - 1; if (yy < 0 || yy >= 28) continue;
      for (int kx = 0; kx < 3; ++kx) {
        int xc = xx + kx - 1; if (xc < 0 || xc >= 28) continue;
        float v = x[b * 784 + yy * 28 + xc];
#pragma unroll
        for (int c = 0; c < 32; ++c) acc[c] += wc1[c * 9 + ky * 3 + kx] * v;
      }
    }
#pragma unroll
    for (int c = 0; c < 32; ++c) {
      tid0[idx * 32 + c] = (_Float16)(wds[c] * xv * m);
      ta[idx * 32 + c]   = (_Float16)(acc[c] * m);
    }
  }
}

__global__ void __launch_bounds__(256) reduce_sum_f16(
    const _Float16* __restrict__ m, long N, float* __restrict__ out)
{
  __shared__ float ls[256];
  int tid = threadIdx.x;
  float s = 0.f;
  for (long idx = (long)blockIdx.x * 256 + tid; idx < N; idx += (long)gridDim.x * 256)
    s += (float)m[idx];
  ls[tid] = s; __syncthreads();
  for (int st = 128; st > 0; st >>= 1) {
    if (tid < st) ls[tid] += ls[tid + st];
    __syncthreads();
  }
  if (tid == 0) atomicAdd(out, ls[0]);
}

__global__ void __launch_bounds__(256) maxpool2_kernel(
    const _Float16* __restrict__ src, _Float16* __restrict__ dst,
    int Bn, int H, int W, int C)
{
  int Ho = H >> 1, Wo = W >> 1;
  long total = (long)Bn * Ho * Wo * C;
  for (long idx = (long)blockIdx.x * 256 + threadIdx.x; idx < total;
       idx += (long)gridDim.x * 256) {
    int c = (int)(idx % C); long t = idx / C;
    int xo = (int)(t % Wo); t /= Wo;
    int yo = (int)(t % Ho); int b = (int)(t / Ho);
    long base = ((long)(b * H + 2 * yo) * W + 2 * xo) * C + c;
    float a = (float)src[base], b2 = (float)src[base + C];
    float c2 = (float)src[base + (long)W * C], d2 = (float)src[base + (long)W * C + C];
    dst[idx] = (_Float16)fmaxf(fmaxf(a, b2), fmaxf(c2, d2));
  }
}

// Per-channel masked sum / sumsq into stats[c], stats[128+c] (f32, atomics)
__global__ void __launch_bounds__(256) bn_stats_kernel(
    const _Float16* __restrict__ x, const _Float16* __restrict__ mask,
    float* __restrict__ stats, long Ntot, int C)
{
  __shared__ float ls[128], lq[128];
  int tid = threadIdx.x;
  for (int c = tid; c < C; c += 256) { ls[c] = 0.f; lq[c] = 0.f; }
  __syncthreads();
  long row0 = (long)blockIdx.x * 128;
  int elems = 128 * C;
  for (int idx = tid; idx < elems; idx += 256) {
    int rr = idx / C, c = idx - rr * C;
    long nn = row0 + rr;
    if (nn < Ntot) {
      float v = (float)x[nn * C + c];
      if (mask) v *= (float)mask[nn];
      atomicAdd(&ls[c], v);
      atomicAdd(&lq[c], v * v);
    }
  }
  __syncthreads();
  for (int c = tid; c < C; c += 256) {
    atomicAdd(&stats[c], ls[c]);
    atomicAdd(&stats[128 + c], lq[c]);
  }
}

__global__ void __launch_bounds__(256) bn_apply_kernel(
    const _Float16* __restrict__ x, _Float16* __restrict__ dst,
    const float* __restrict__ stats, const float* __restrict__ cntp, float cntc,
    const _Float16* __restrict__ mask, const _Float16* __restrict__ addsrc,
    long Ntot, int C, int relu)
{
  float cnt = cntp ? fmaxf(*cntp, 1.f) : cntc;
  long total = Ntot * C;
  for (long idx = (long)blockIdx.x * 256 + threadIdx.x; idx < total;
       idx += (long)gridDim.x * 256) {
    int c = (int)(idx % C);
    long nn = idx / C;
    float mean = stats[c] / cnt;
    float var  = stats[128 + c] / cnt - mean * mean;
    float inv  = rsqrtf(var + 1e-5f);
    float v = ((float)x[idx] - mean) * inv;
    if (mask)   v *= (float)mask[nn];
    if (addsrc) v += (float)addsrc[idx];
    if (relu)   v = fmaxf(v, 0.f);
    dst[idx] = (_Float16)v;
  }
}

__global__ void __launch_bounds__(256) copy_ch_kernel(
    const _Float16* __restrict__ src, _Float16* __restrict__ dst,
    long Ntot, int Cs, int Cd, int off)
{
  long total = Ntot * Cs;
  for (long idx = (long)blockIdx.x * 256 + threadIdx.x; idx < total;
       idx += (long)gridDim.x * 256) {
    long n = idx / Cs; int c = (int)(idx % Cs);
    dst[n * Cd + off + c] = src[idx];
  }
}

// f32 [O][I][P] (OIHW flat) -> f16 [P][O][I]
__global__ void repack_conv_kernel(const float* __restrict__ w, _Float16* __restrict__ o_,
                                   int O, int I, int P)
{
  int idx = blockIdx.x * 256 + threadIdx.x;
  int tot = O * I * P;
  if (idx >= tot) return;
  int p = idx % P; int t = idx / P; int i = t % I; int oo = t / I;
  o_[((long)p * O + oo) * I + i] = (_Float16)w[((long)oo * I + i) * P + p];
}

// deconv f32 [I(cin)][O(cout)][2][2] -> f16 [4(pq)][O][I]
__global__ void repack_deconv_kernel(const float* __restrict__ w, _Float16* __restrict__ o_,
                                     int O, int I)
{
  int idx = blockIdx.x * 256 + threadIdx.x;
  int tot = O * I * 4;
  if (idx >= tot) return;
  int p = idx & 3; int t = idx >> 2; int i = t % I; int oo = t / I;
  o_[((long)p * O + oo) * I + i] = (_Float16)w[((long)i * O + oo) * 4 + p];
}

// mean-pool over 784 spatial, 10-way head, log_softmax. One block per batch elem.
__global__ void __launch_bounds__(256) head_kernel(
    const _Float16* __restrict__ y, const float* __restrict__ wh,
    const float* __restrict__ bh, float* __restrict__ out)
{
  __shared__ float ps[32];
  int tid = threadIdx.x; int b = blockIdx.x;
  if (tid < 32) ps[tid] = 0.f;
  __syncthreads();
  const _Float16* yb = y + (long)b * 784 * 32;
  float s = 0.f;
  for (int idx = tid; idx < 784 * 32; idx += 256) s += (float)yb[idx];
  atomicAdd(&ps[tid & 31], s);
  __syncthreads();
  if (tid == 0) {
    float logits[10]; float mx = -1e30f;
    for (int o = 0; o < 10; ++o) {
      float l = bh[o];
      for (int c = 0; c < 32; ++c) l += (ps[c] / 784.f) * wh[o * 32 + c];
      logits[o] = l; mx = fmaxf(mx, l);
    }
    float s2 = 0.f;
    for (int o = 0; o < 10; ++o) s2 += __expf(logits[o] - mx);
    float lse = mx + __logf(s2);
    for (int o = 0; o < 10; ++o) out[b * 10 + o] = logits[o] - lse;
  }
}

// --------------------------- host-side dispatch ----------------------------

static void gemm_dispatch(const _Float16* src, const _Float16* wt, _Float16* dst,
                          const float* bias, const _Float16* mask, const _Float16* resid,
                          int Bn, int H, int W, int Cin, int Cout, int npos, int act,
                          hipStream_t stream)
{
  const long Nt = (long)Bn * H * W;
  const int MT = Cout / 16, KC = Cin / 32;
#define GCASE(mt, kc, p)                                                              \
  if (MT == (mt) && KC == (kc) && npos == (p)) {                                      \
    constexpr int NGv = ((mt) < 8) ? (8 / (mt)) : 1;                                  \
    conv_gemm_wmma<(mt), (kc), (p)>                                                   \
        <<<dim3((unsigned)(Nt / (16 * NGv))), dim3(256), 0, stream>>>(                \
            src, wt, dst, bias, mask, resid, Bn, H, W, act);                          \
    return;                                                                           \
  }
  GCASE(2, 1, 9)   // e1 conv2, d0 conv2
  GCASE(2, 2, 1)   // d0 downsample
  GCASE(2, 2, 9)   // d0 conv1
  GCASE(4, 1, 1)   // e2 downsample
  GCASE(4, 1, 9)   // e2 conv1
  GCASE(4, 2, 9)   // e2 conv2, d1 conv2
  GCASE(4, 4, 1)   // post, d1 downsample
  GCASE(4, 4, 9)   // d1 conv1
  GCASE(8, 2, 1)   // pre
  GCASE(8, 4, 1)   // attn proj
  GCASE(8, 8, 1)   // mlp2
  GCASE(16, 4, 1)  // mlp1
  GCASE(24, 4, 1)  // qkv
#undef GCASE
}

// ===========================================================================

extern "C" void kernel_launch(void* const* d_in, const int* in_sizes, int n_in,
                              void* d_out, int out_size, void* d_ws, size_t ws_size,
                              hipStream_t stream)
{
  (void)in_sizes; (void)n_in; (void)out_size; (void)ws_size;
  const float* x       = (const float*)d_in[0];
  const float* w_e1_ds = (const float*)d_in[1];
  const float* w_e1_c1 = (const float*)d_in[2];
  const float* w_e1_c2 = (const float*)d_in[3];
  const float* w_e2_ds = (const float*)d_in[4];
  const float* w_e2_c1 = (const float*)d_in[5];
  const float* w_e2_c2 = (const float*)d_in[6];
  const float* w_pre   = (const float*)d_in[7];
  const float* b_pre   = (const float*)d_in[8];
  const float* w_qkv   = (const float*)d_in[9];
  const float* b_qkv   = (const float*)d_in[10];
  const float* w_o     = (const float*)d_in[11];
  const float* b_o     = (const float*)d_in[12];
  const float* w_m1    = (const float*)d_in[13];
  const float* b_m1    = (const float*)d_in[14];
  const float* w_m2    = (const float*)d_in[15];
  const float* b_m2    = (const float*)d_in[16];
  const float* w_post  = (const float*)d_in[17];
  const float* b_post  = (const float*)d_in[18];
  const float* w_up1   = (const float*)d_in[19];
  const float* w_d1_ds = (const float*)d_in[20];
  const float* w_d1_c1 = (const float*)d_in[21];
  const float* w_d1_c2 = (const float*)d_in[22];
  const float* w_up0   = (const float*)d_in[23];
  const float* w_d0_ds = (const float*)d_in[24];
  const float* w_d0_c1 = (const float*)d_in[25];
  const float* w_d0_c2 = (const float*)d_in[26];
  const float* w_head  = (const float*)d_in[27];
  const float* b_head  = (const float*)d_in[28];

  const int B = 2048;
  const long N0 = (long)B * 784, N1 = (long)B * 196, N2 = (long)B * 49;

  char* base = (char*)d_ws; size_t off = 0;
  auto allocH = [&](size_t n) -> _Float16* {
    off = (off + 255) & ~(size_t)255;
    _Float16* p = (_Float16*)(base + off); off += n * sizeof(_Float16); return p;
  };
  auto allocF = [&](size_t n) -> float* {
    off = (off + 255) & ~(size_t)255;
    float* p = (float*)(base + off); off += n * sizeof(float); return p;
  };

  _Float16 *m0 = allocH(N0), *m1 = allocH(N1), *m2 = allocH(N2);
  _Float16 *B1 = allocH(N0 * 32), *B2 = allocH(N0 * 32), *B3 = allocH(N0 * 32);
  _Float16 *B8 = allocH(N0 * 64);
  _Float16 *B4 = allocH(N1 * 32);
  _Float16 *B5 = allocH(N1 * 64), *B6 = allocH(N1 * 64), *B7 = allocH(N1 * 64);
  _Float16 *S1 = allocH(N2 * 64), *S2 = allocH(N2 * 128), *S3 = allocH(N2 * 128);
  _Float16 *S5 = allocH(N2 * 128), *S4 = allocH(N2 * 384), *S6 = allocH(N2 * 256);
  _Float16 *we1c2 = allocH(9 * 32 * 32);
  _Float16 *we2ds = allocH(64 * 32), *we2c1 = allocH(9 * 64 * 32), *we2c2 = allocH(9 * 64 * 64);
  _Float16 *wpre = allocH(128 * 64), *wqkv = allocH(384 * 128), *wo16 = allocH(128 * 128);
  _Float16 *wm1 = allocH(256 * 128), *wm2 = allocH(128 * 256), *wpost = allocH(64 * 128);
  _Float16 *wup1 = allocH(4 * 64 * 64), *wd1ds = allocH(64 * 128);
  _Float16 *wd1c1 = allocH(9 * 64 * 128), *wd1c2 = allocH(9 * 64 * 64);
  _Float16 *wup0 = allocH(4 * 32 * 64), *wd0ds = allocH(32 * 64);
  _Float16 *wd0c1 = allocH(9 * 32 * 64), *wd0c2 = allocH(9 * 32 * 32);
  float* stats = allocF(16 * 256);
  float* cnts  = allocF(8);
  auto st = [&](int s) -> float* { return stats + s * 256; };

  auto gemm = [&](const _Float16* src, const _Float16* wt, _Float16* dst,
                  const float* bias, const _Float16* mask, const _Float16* resid,
                  int Bn, int H, int W, int Cin, int Cout, int npos, int act) {
    gemm_dispatch(src, wt, dst, bias, mask, resid, Bn, H, W, Cin, Cout, npos, act, stream);
  };
  auto stats_k = [&](const _Float16* xx, const _Float16* mk, float* slot, long Nt, int C) {
    bn_stats_kernel<<<dim3((unsigned)((Nt + 127) / 128)), 256, 0, stream>>>(xx, mk, slot, Nt, C);
  };
  auto bn = [&](const _Float16* xx, _Float16* dd, float* slot, const float* cp, float cc,
                const _Float16* mk, const _Float16* ad, long Nt, int C, int relu) {
    bn_apply_kernel<<<dim3(4096), 256, 0, stream>>>(xx, dd, slot, cp, cc, mk, ad, Nt, C, relu);
  };
  auto rp = [&](const float* w, _Float16* o_, int O, int I, int P) {
    int tot = O * I * P;
    repack_conv_kernel<<<dim3((tot + 255) / 256), 256, 0, stream>>>(w, o_, O, I, P);
  };
  auto pool = [&](const _Float16* s, _Float16* d, int Bn, int H, int W, int C) {
    maxpool2_kernel<<<dim3(4096), 256, 0, stream>>>(s, d, Bn, H, W, C);
  };
  auto copych = [&](const _Float16* s, _Float16* d, long Nt, int Cs, int Cd, int o_) {
    copy_ch_kernel<<<dim3(4096), 256, 0, stream>>>(s, d, Nt, Cs, Cd, o_);
  };

  // ---- weight repack (f32 -> f16, WMMA-friendly [pos][O][I]) ----
  rp(w_e1_c2, we1c2, 32, 32, 9);
  rp(w_e2_ds, we2ds, 64, 32, 1); rp(w_e2_c1, we2c1, 64, 32, 9); rp(w_e2_c2, we2c2, 64, 64, 9);
  rp(w_pre, wpre, 128, 64, 1);   rp(w_qkv, wqkv, 384, 128, 1);  rp(w_o, wo16, 128, 128, 1);
  rp(w_m1, wm1, 256, 128, 1);    rp(w_m2, wm2, 128, 256, 1);    rp(w_post, wpost, 64, 128, 1);
  rp(w_d1_ds, wd1ds, 64, 128, 1); rp(w_d1_c1, wd1c1, 64, 128, 9); rp(w_d1_c2, wd1c2, 64, 64, 9);
  rp(w_d0_ds, wd0ds, 32, 64, 1);  rp(w_d0_c1, wd0c1, 32, 64, 9);  rp(w_d0_c2, wd0c2, 32, 32, 9);
  repack_deconv_kernel<<<dim3((64 * 64 * 4 + 255) / 256), 256, 0, stream>>>(w_up1, wup1, 64, 64);
  repack_deconv_kernel<<<dim3((32 * 64 * 4 + 255) / 256), 256, 0, stream>>>(w_up0, wup0, 32, 64);

  zero_f32_kernel<<<dim3((16 * 256 + 255) / 256), 256, 0, stream>>>(stats, 16 * 256);
  zero_f32_kernel<<<dim3(1), 256, 0, stream>>>(cnts, 8);

  // ---- encoder level 0 (28x28) ----
  front_kernel<<<dim3(6272), 256, 0, stream>>>(x, w_e1_ds, w_e1_c1, m0, B1, B2, N0);
  reduce_sum_f16<<<dim3(1024), 256, 0, stream>>>(m0, N0, cnts + 0);
  pool(m0, m1, B, 28, 28, 1);
  pool(m1, m2, B, 14, 14, 1);
  reduce_sum_f16<<<dim3(512), 256, 0, stream>>>(m1, N1, cnts + 1);
  reduce_sum_f16<<<dim3(256), 256, 0, stream>>>(m2, N2, cnts + 2);
  stats_k(B1, m0, st(0), N0, 32); stats_k(B2, m0, st(1), N0, 32);
  bn(B1, B1, st(0), cnts + 0, 0.f, m0, nullptr, N0, 32, 0);            // ident0
  bn(B2, B2, st(1), cnts + 0, 0.f, m0, nullptr, N0, 32, 1);            // a0
  gemm(B2, we1c2, B3, nullptr, m0, nullptr, B, 28, 28, 32, 32, 9, 0);  // conv2*m
  stats_k(B3, m0, st(2), N0, 32);
  bn(B3, B2, st(2), cnts + 0, 0.f, m0, B1, N0, 32, 1);                 // e1 -> B2
  pool(B2, B4, B, 28, 28, 32);                                         // p1

  // ---- encoder level 1 (14x14) ----
  gemm(B4, we2ds, B5, nullptr, m1, nullptr, B, 14, 14, 32, 64, 1, 0);
  gemm(B4, we2c1, B6, nullptr, m1, nullptr, B, 14, 14, 32, 64, 9, 0);
  stats_k(B5, m1, st(3), N1, 64); stats_k(B6, m1, st(4), N1, 64);
  bn(B5, B5, st(3), cnts + 1, 0.f, m1, nullptr, N1, 64, 0);
  bn(B6, B6, st(4), cnts + 1, 0.f, m1, nullptr, N1, 64, 1);
  gemm(B6, we2c2, B7, nullptr, m1, nullptr, B, 14, 14, 64, 64, 9, 0);
  stats_k(B7, m1, st(5), N1, 64);
  bn(B7, B6, st(5), cnts + 1, 0.f, m1, B5, N1, 64, 1);                 // e2 -> B6
  pool(B6, S1, B, 14, 14, 64);                                         // p2

  // ---- bottleneck / attention (7x7 tokens) ----
  gemm(S1, wpre, S2, b_pre, nullptr, nullptr, B, 7, 7, 64, 128, 1, 0); // d
  stats_k(S2, nullptr, st(6), N2, 128);
  bn(S2, S3, st(6), nullptr, (float)N2, nullptr, nullptr, N2, 128, 0); // xn
  gemm(S3, wqkv, S4, b_qkv, nullptr, nullptr, B, 7, 7, 128, 384, 1, 0);
  attn_kernel<<<dim3(B * 4), 256, 0, stream>>>(S4, S5);                // o
  gemm(S5, wo16, S5, b_o, nullptr, S3, B, 7, 7, 128, 128, 1, 0);       // x2 = xn + proj(o)
  gemm(S5, wm1, S6, b_m1, nullptr, nullptr, B, 7, 7, 128, 256, 1, 2);  // gelu mlp1
  gemm(S6, wm2, S3, b_m2, nullptr, S5, B, 7, 7, 256, 128, 1, 0);       // x2 + mlp -> S3
  gemm(S3, wpost, S1, b_post, m2, nullptr, B, 7, 7, 128, 64, 1, 0);    // xs -> S1

  // ---- decoder level 1 ----
  deconv2x2_wmma<4, 2><<<dim3((unsigned)(N2 / 16)), 256, 0, stream>>>(S1, wup1, B7, m1, B, 7, 7);
  stats_k(B7, m1, st(7), N1, 64);
  bn(B7, B7, st(7), cnts + 1, 0.f, m1, nullptr, N1, 64, 1);            // y1
  _Float16* D1 = B3;                                                   // [B,14,14,128]
  copych(B7, D1, N1, 64, 128, 0);
  copych(B6, D1, N1, 64, 128, 64);
  gemm(D1, wd1ds, B5, nullptr, m1, nullptr, B, 14, 14, 128, 64, 1, 0);
  gemm(D1, wd1c1, B7, nullptr, m1, nullptr, B, 14, 14, 128, 64, 9, 0);
  stats_k(B5, m1, st(8), N1, 64); stats_k(B7, m1, st(9), N1, 64);
  bn(B5, B5, st(8), cnts + 1, 0.f, m1, nullptr, N1, 64, 0);
  bn(B7, B7, st(9), cnts + 1, 0.f, m1, nullptr, N1, 64, 1);
  gemm(B7, wd1c2, B6, nullptr, m1, nullptr, B, 14, 14, 64, 64, 9, 0);
  stats_k(B6, m1, st(10), N1, 64);
  bn(B6, B7, st(10), cnts + 1, 0.f, m1, B5, N1, 64, 1);                // d1 -> B7

  // ---- decoder level 0 ----
  deconv2x2_wmma<2, 2><<<dim3((unsigned)(N1 / 16)), 256, 0, stream>>>(B7, wup0, B3, m0, B, 14, 14);
  stats_k(B3, m0, st(11), N0, 32);
  bn(B3, B3, st(11), cnts + 0, 0.f, m0, nullptr, N0, 32, 1);           // y0
  _Float16* D0 = B8;                                                   // [B,28,28,64]
  copych(B3, D0, N0, 32, 64, 0);
  copych(B2, D0, N0, 32, 64, 32);
  gemm(D0, wd0ds, B1, nullptr, m0, nullptr, B, 28, 28, 64, 32, 1, 0);
  gemm(D0, wd0c1, B3, nullptr, m0, nullptr, B, 28, 28, 64, 32, 9, 0);
  stats_k(B1, m0, st(12), N0, 32); stats_k(B3, m0, st(13), N0, 32);
  bn(B1, B1, st(12), cnts + 0, 0.f, m0, nullptr, N0, 32, 0);
  bn(B3, B3, st(13), cnts + 0, 0.f, m0, nullptr, N0, 32, 1);
  gemm(B3, wd0c2, B2, nullptr, m0, nullptr, B, 28, 28, 32, 32, 9, 0);
  stats_k(B2, m0, st(14), N0, 32);
  bn(B2, B3, st(14), cnts + 0, 0.f, m0, B1, N0, 32, 1);                // d0 -> B3

  // ---- head ----
  head_kernel<<<dim3(B), 256, 0, stream>>>(B3, w_head, b_head, (float*)d_out);
}
